// RGCNEncoder_6012954214569
// MI455X (gfx1250) — compile-verified, matrix operand order checked by reading, not run on hardware
//
#include <hip/hip_runtime.h>
#include <hip/hip_bf16.h>

typedef __bf16 v16bf __attribute__((ext_vector_type(16)));
typedef __bf16 v8bf  __attribute__((ext_vector_type(8)));
typedef __bf16 v4bf  __attribute__((ext_vector_type(4)));
typedef float  v8f   __attribute__((ext_vector_type(8)));

#define N_ENT   50000
#define N_REL   16
#define N_BASES 8
#define H_DIM   128
#define KTOT    2176        // 16*128 relation part + 128 self-loop part
#define KC      32          // K chunk (one bf16 WMMA K-slab)
#define NCHUNK  (KTOT / KC) // 68
#define APITCH  40          // LDS pitch in bf16 elements (16B aligned, conflict-free)

// ---------------------------------------------------------------------------
// Build stacked, transposed B matrix in bf16 hi/lo split precision.
// B[k][o]: k in [0,2048) -> W_r[i][o] = sum_b w_comp[r,b]*basis[b,i,o]
//          k in [2048,2176) -> loop_w[k-2048][o]
// Stored transposed: BT[o*KTOT + k]  (contiguous K-runs for the GEMM).
// ---------------------------------------------------------------------------
__global__ void build_B(const float* __restrict__ basis,
                        const float* __restrict__ w_comp,
                        const float* __restrict__ loop_w,
                        __bf16* __restrict__ BhT,
                        __bf16* __restrict__ BlT) {
    int idx = blockIdx.x * blockDim.x + threadIdx.x;   // over 128 * KTOT
    if (idx >= H_DIM * KTOT) return;
    int o = idx / KTOT;
    int k = idx - o * KTOT;
    float v;
    if (k < N_REL * H_DIM) {
        int r = k >> 7;
        int i = k & 127;
        float acc = 0.f;
#pragma unroll
        for (int b = 0; b < N_BASES; ++b)
            acc += w_comp[r * N_BASES + b] * basis[((size_t)b * H_DIM + i) * H_DIM + o];
        v = acc;
    } else {
        v = loop_w[(size_t)(k - N_REL * H_DIM) * H_DIM + o];
    }
    __bf16 hi = (__bf16)v;               // RNE
    float  lo = v - (float)hi;
    BhT[idx] = hi;
    BlT[idx] = (__bf16)lo;
}

// ---------------------------------------------------------------------------
// Zero the aggregation buffer (float4 grid-stride).
// ---------------------------------------------------------------------------
__global__ void zero_f4(float4* __restrict__ p, size_t n4) {
    size_t i = (size_t)blockIdx.x * blockDim.x + threadIdx.x;
    size_t stride = (size_t)gridDim.x * blockDim.x;
    float4 z = make_float4(0.f, 0.f, 0.f, 0.f);
    for (; i < n4; i += stride) p[i] = z;
}

// ---------------------------------------------------------------------------
// Edge scatter: one wave (32 lanes) per edge; each lane handles 4 floats.
// Edge indices are wave-uniform -> force scalar loads via readfirstlane.
// ---------------------------------------------------------------------------
__global__ void scatter_edges(const float* __restrict__ h,
                              const int* __restrict__ src,
                              const int* __restrict__ dst,
                              const int* __restrict__ rel,
                              float* __restrict__ agg, int nEdges) {
    int t = blockIdx.x * blockDim.x + threadIdx.x;
    int e = t >> 5;
    int lane = t & 31;
    if (e >= nEdges) return;
    e = __builtin_amdgcn_readfirstlane(e);
    int s = src[e];
    int d = dst[e];
    int r = rel[e];
    const float4 v = *(const float4*)(h + (size_t)s * H_DIM + lane * 4);
    float* base = agg + ((size_t)d * N_REL + r) * H_DIM + lane * 4;
    atomicAdd(base + 0, v.x);
    atomicAdd(base + 1, v.y);
    atomicAdd(base + 2, v.z);
    atomicAdd(base + 3, v.w);
}

// ---------------------------------------------------------------------------
// GEMM helpers
// ---------------------------------------------------------------------------
__device__ __forceinline__ v16bf frag16(const __bf16* p) {
    v8bf lo = *(const v8bf*)(p);
    v8bf hi = *(const v8bf*)(p + 16);
    return __builtin_shufflevector(lo, hi, 0, 1, 2, 3, 4, 5, 6, 7,
                                          8, 9, 10, 11, 12, 13, 14, 15);
}

// Per-lane async copy of one 16B B-tile slot, global -> LDS (no VGPR data path).
__device__ __forceinline__ void async_b128(const __bf16* lds_dst, const __bf16* gsrc) {
    unsigned int       lds_off = (unsigned int)(uintptr_t)lds_dst;   // low 32 = LDS offset
    unsigned long long gaddr   = (unsigned long long)(uintptr_t)gsrc;
    asm volatile("global_load_async_to_lds_b128 %0, %1, off"
                 :: "v"(lds_off), "v"(gaddr) : "memory");
}

__device__ __forceinline__ void wait_async0() {
    asm volatile("s_wait_asynccnt 0x0" ::: "memory");
}

__device__ __forceinline__ void loadA_regs(float4* areg,
                                           const float* __restrict__ agg,
                                           const float* __restrict__ hsrc,
                                           int nBase, int N, int kglob, int tid) {
#pragma unroll
    for (int p = 0; p < 4; ++p) {
        int linear = p * 256 + tid;       // 1024 float4 slots: 128 rows x 8
        int row = linear >> 3;
        int kq  = (linear & 7) * 4;
        float4 v = make_float4(0.f, 0.f, 0.f, 0.f);
        int n = nBase + row;
        if (n < N) {
            if (kglob < N_REL * H_DIM)
                v = *(const float4*)(agg + (size_t)n * (N_REL * H_DIM) + kglob + kq);
            else
                v = *(const float4*)(hsrc + (size_t)n * H_DIM + (kglob - N_REL * H_DIM) + kq);
        }
        areg[p] = v;
    }
}

__device__ __forceinline__ void storeA_lds(const float4* areg,
                                           __bf16* Ah, __bf16* Al, int tid) {
#pragma unroll
    for (int p = 0; p < 4; ++p) {
        int linear = p * 256 + tid;
        int row = linear >> 3;
        int kq  = (linear & 7) * 4;
        float f[4] = {areg[p].x, areg[p].y, areg[p].z, areg[p].w};
        v4bf hv, lv;
#pragma unroll
        for (int j = 0; j < 4; ++j) {
            __bf16 hb = (__bf16)f[j];
            hv[j] = hb;
            lv[j] = (__bf16)(f[j] - (float)hb);
        }
        *(v4bf*)&Ah[row * APITCH + kq] = hv;
        *(v4bf*)&Al[row * APITCH + kq] = lv;
    }
}

__device__ __forceinline__ void issueB_async(__bf16* Bh, __bf16* Bl,
                                             const __bf16* __restrict__ BhT,
                                             const __bf16* __restrict__ BlT,
                                             int kglob, int tid) {
#pragma unroll
    for (int p = 0; p < 2; ++p) {
        int linear = p * 256 + tid;       // 512 v8bf slots: 128 cols x 4
        int col = linear >> 2;
        int kq  = (linear & 3) * 8;
        async_b128(&Bh[col * APITCH + kq], BhT + (size_t)col * KTOT + kglob + kq);
        async_b128(&Bl[col * APITCH + kq], BlT + (size_t)col * KTOT + kglob + kq);
    }
}

// ---------------------------------------------------------------------------
// Main fused GEMM: out[n,:] = relu( [agg[n,:] | h[n,:]] @ B + bias )
// Double-buffered LDS; B staged with global_load_async_to_lds_b128 (ASYNCcnt),
// A prefetched into registers and split fp32 -> bf16 hi/lo.
// Split-precision: acc += Ahi*Bhi + Ahi*Blo + Alo*Bhi (fp32 accumulate).
// ---------------------------------------------------------------------------
__global__ __launch_bounds__(256) void rgcn_gemm(
    const float* __restrict__ agg, const float* __restrict__ hsrc,
    const __bf16* __restrict__ BhT, const __bf16* __restrict__ BlT,
    const float* __restrict__ bias, float* __restrict__ out, int N) {

    __shared__ __bf16 Ah[2][128 * APITCH];
    __shared__ __bf16 Al[2][128 * APITCH];
    __shared__ __bf16 Bh[2][128 * APITCH];   // transposed: [col][k]
    __shared__ __bf16 Bl[2][128 * APITCH];

    const int tid  = threadIdx.x;
    const int wave = tid >> 5;
    const int lane = tid & 31;
    const int half = lane >> 4;
    const int l15  = lane & 15;
    const int kb   = half * 8;            // K-sub-slab base for 16-bit fragments
    const int nBase = blockIdx.x * 128;
    const int mrow  = wave * 16 + l15;

    v8f acc[8];
#pragma unroll
    for (int i = 0; i < 8; ++i) acc[i] = (v8f)0.f;

    // ---- prologue: prefetch chunk 0 ----
    float4 areg[4];
    loadA_regs(areg, agg, hsrc, nBase, N, 0, tid);
    issueB_async(Bh[0], Bl[0], BhT, BlT, 0, tid);

    for (int c = 0; c < NCHUNK; ++c) {
        const int cur = c & 1;
        const int nxt = cur ^ 1;

        // convert + commit A chunk c to LDS; wait this thread's async B(c)
        storeA_lds(areg, Ah[cur], Al[cur], tid);
        wait_async0();
        __syncthreads();      // buffers for chunk c are now fully visible

        // prefetch chunk c+1 (safe: everyone finished compute on buffer nxt)
        if (c + 1 < NCHUNK) {
            loadA_regs(areg, agg, hsrc, nBase, N, (c + 1) * KC, tid);
            issueB_async(Bh[nxt], Bl[nxt], BhT, BlT, (c + 1) * KC, tid);
        }

        // ---- compute chunk c: 8 col-tiles x 3 split-precision WMMAs ----
        v16bf ah = frag16(&Ah[cur][mrow * APITCH + kb]);
        v16bf al = frag16(&Al[cur][mrow * APITCH + kb]);
#pragma unroll
        for (int nt = 0; nt < 8; ++nt) {
            int col = nt * 16 + l15;
            v16bf bh = frag16(&Bh[cur][col * APITCH + kb]);
            v16bf bl = frag16(&Bl[cur][col * APITCH + kb]);
            acc[nt] = __builtin_amdgcn_wmma_f32_16x16x32_bf16(
                false, ah, false, bh, (short)0, acc[nt], false, false);
            acc[nt] = __builtin_amdgcn_wmma_f32_16x16x32_bf16(
                false, ah, false, bl, (short)0, acc[nt], false, false);
            acc[nt] = __builtin_amdgcn_wmma_f32_16x16x32_bf16(
                false, al, false, bh, (short)0, acc[nt], false, false);
        }
    }

    // ---- epilogue: bias + relu, C layout: VGPR j -> M = j + 8*(lane>>4) ----
    const int rowoff = half * 8;
#pragma unroll
    for (int nt = 0; nt < 8; ++nt) {
        int o = nt * 16 + l15;
        float bv = bias[o];
#pragma unroll
        for (int j = 0; j < 8; ++j) {
            int m = nBase + wave * 16 + rowoff + j;
            if (m < N)
                out[(size_t)m * H_DIM + o] = fmaxf(acc[nt][j] + bv, 0.f);
        }
    }
}

// ---------------------------------------------------------------------------
extern "C" void kernel_launch(void* const* d_in, const int* in_sizes, int n_in,
                              void* d_out, int out_size, void* d_ws, size_t ws_size,
                              hipStream_t stream) {
    const float* emb     = (const float*)d_in[0];
    const float* basis1  = (const float*)d_in[1];
    const float* w_comp1 = (const float*)d_in[2];
    const float* loop_w1 = (const float*)d_in[3];
    const float* bias1   = (const float*)d_in[4];
    const float* basis2  = (const float*)d_in[5];
    const float* w_comp2 = (const float*)d_in[6];
    const float* loop_w2 = (const float*)d_in[7];
    const float* bias2   = (const float*)d_in[8];
    const int*   src     = (const int*)d_in[9];
    const int*   dst     = (const int*)d_in[10];
    const int*   rel     = (const int*)d_in[11];
    float* out = (float*)d_out;

    const int nEdges = in_sizes[9];
    const int N = in_sizes[0] / H_DIM;

    // Workspace layout
    char* ws = (char*)d_ws;
    const size_t aggBytes = (size_t)N_ENT * N_REL * H_DIM * sizeof(float); // 409.6 MB
    const size_t h1Bytes  = (size_t)N_ENT * H_DIM * sizeof(float);         // 25.6 MB
    const size_t bBytes   = (size_t)KTOT * H_DIM * sizeof(__bf16);         // 557 KB each
    float*  agg = (float*)ws;
    float*  h1  = (float*)(ws + aggBytes);
    __bf16* B1h = (__bf16*)(ws + aggBytes + h1Bytes);
    __bf16* B1l = (__bf16*)((char*)B1h + bBytes);
    __bf16* B2h = (__bf16*)((char*)B1l + bBytes);
    __bf16* B2l = (__bf16*)((char*)B2h + bBytes);

    const size_t agg4 = (size_t)N_ENT * N_REL * H_DIM / 4;
    const int bBlocks    = (H_DIM * KTOT + 255) / 256;
    const int zeroBlocks = 8192;
    const int scatBlocks = (nEdges * 32 + 255) / 256;
    const int gemmBlocks = (N + 127) / 128;

    // Precompute split-precision B matrices for both layers
    build_B<<<bBlocks, 256, 0, stream>>>(basis1, w_comp1, loop_w1, B1h, B1l);
    build_B<<<bBlocks, 256, 0, stream>>>(basis2, w_comp2, loop_w2, B2h, B2l);

    // ---- Layer 1 ----
    zero_f4<<<zeroBlocks, 256, 0, stream>>>((float4*)agg, agg4);
    scatter_edges<<<scatBlocks, 256, 0, stream>>>(emb, src, dst, rel, agg, nEdges);
    rgcn_gemm<<<gemmBlocks, 256, 0, stream>>>(agg, emb, B1h, B1l, bias1, h1, N);

    // ---- Layer 2 ----
    zero_f4<<<zeroBlocks, 256, 0, stream>>>((float4*)agg, agg4);
    scatter_edges<<<scatBlocks, 256, 0, stream>>>(h1, src, dst, rel, agg, nEdges);
    rgcn_gemm<<<gemmBlocks, 256, 0, stream>>>(agg, h1, B2h, B2l, bias2, out, N);
}
